// MultiHeadAttention_33638183862297
// MI455X (gfx1250) — matrix operand order repaired and checked by reading, not compiled, not run on hardware
//
#include <hip/hip_runtime.h>

#define DEV __device__ __forceinline__

typedef _Float16 h16;
typedef __attribute__((ext_vector_type(16))) _Float16 v16h;
typedef __attribute__((ext_vector_type(8)))  _Float16 v8h;
typedef __attribute__((ext_vector_type(8)))  float    v8f;
typedef __attribute__((ext_vector_type(4)))  float    v4f;

constexpr int D_MODEL = 1024;
constexpr int N_HEAD  = 16;
constexpr int D_K     = 64;
constexpr int SEQ     = 2048;
constexpr int BATCH   = 2;
constexpr int ROWS    = BATCH * SEQ;   // 4096
constexpr int LDSTR   = 40;            // LDS row stride in f16 (80B, keeps 16B alignment)

DEV v8f vzero() {
  v8f z;
#pragma unroll
  for (int e = 0; e < 8; ++e) z[e] = 0.0f;
  return z;
}

DEV v8f wmma_f16(v16h a, v16h b, v8f c) {
  // D = A(16x32 f16) * B(32x16 f16) + C(16x16 f32)
  return __builtin_amdgcn_wmma_f32_16x16x32_f16(false, a, false, b, (short)0, c, false, false);
}

DEV v16h pack16(v8h lo, v8h hi) {
  v16h f;
#pragma unroll
  for (int i = 0; i < 8; ++i) { f[i] = lo[i]; f[8 + i] = hi[i]; }
  return f;
}

// A fragment (16x32 f16) from row-major f16 src; lane row = row0 + lane%16,
// K striping: lanes<16 hold K {0..7, 16..23}, lanes>=16 hold K {8..15, 24..31}.
DEV v16h load_a16(const h16* src, int ld, int row0, int k0, int lane) {
  const h16* p = src + (size_t)(row0 + (lane & 15)) * ld + k0 + ((lane >> 4) ? 8 : 0);
  return pack16(*(const v8h*)p, *(const v8h*)(p + 16));
}

// Same A fragment but from row-major f32 src with on-the-fly f32->f16 convert.
DEV v16h load_a32(const float* src, int ld, int row0, int k0, int lane) {
  const float* p = src + (size_t)(row0 + (lane & 15)) * ld + k0 + ((lane >> 4) ? 8 : 0);
  v4f a0 = *(const v4f*)p,        a1 = *(const v4f*)(p + 4);
  v4f b0 = *(const v4f*)(p + 16), b1 = *(const v4f*)(p + 20);
  v16h f;
#pragma unroll
  for (int i = 0; i < 4; ++i) {
    f[i]      = (h16)a0[i];  f[4 + i]  = (h16)a1[i];
    f[8 + i]  = (h16)b0[i];  f[12 + i] = (h16)b1[i];
  }
  return f;
}

// B fragment (32x16 f16) taken from B^T stored row-major (rows = N, contiguous K).
// Lanes 0..15 hold K {0..7, 16..23} of column N=lane, lanes 16..31 hold K {8..15, 24..31}.
DEV v16h load_b16(const h16* srcT, int ld, int col0, int k0, int lane) {
  const h16* p = srcT + (size_t)(col0 + (lane & 15)) * ld + k0 + ((lane >> 4) ? 8 : 0);
  return pack16(*(const v8h*)p, *(const v8h*)(p + 16));   // two 16B loads, 16B aligned
}

// A fragment from an LDS tile (64 x 32 attn scores, row stride LDSTR halves).
DEV v16h load_a_lds(const h16* sm, int row0, int lane) {
  const h16* p = sm + (row0 + (lane & 15)) * LDSTR + ((lane >> 4) ? 8 : 0);
  return pack16(*(const v8h*)p, *(const v8h*)(p + 16));
}

// ---------------------------------------------------------------------------
// Kernel 0: W^T convert to f16 for all four weights (B-fragments want N-major).
__global__ __launch_bounds__(256) void wcvt_kernel(const float* Wq, const float* Wk,
                                                   const float* Wv, const float* Wo, h16* WT) {
  const float* W = blockIdx.z == 0 ? Wq : blockIdx.z == 1 ? Wk : blockIdx.z == 2 ? Wv : Wo;
  h16* out = WT + (size_t)blockIdx.z * D_MODEL * D_MODEL;
  int idx = blockIdx.x * blockDim.x + threadIdx.x;   // over D*D
  int n = idx >> 10, k = idx & 1023;
  out[idx] = (h16)W[k * D_MODEL + n];                // out[n][k] = W[k][n]
}

// ---------------------------------------------------------------------------
// Kernel 1: fused QKV projection. z selects q/k/v. Wave owns a 64x64 tile.
// Q,K stored [bh][s][64] f16; V stored transposed [bh][64][t] f16.
__global__ __launch_bounds__(128) void qkv_proj_kernel(
    const float* xq, const float* xk, const float* xv, const h16* WT,
    const float* bq, const float* bk, const float* bv,
    h16* Qh, h16* Kh, h16* VhT) {
  const int lane = threadIdx.x & 31;
  const int wave = threadIdx.x >> 5;
  const int z    = blockIdx.z;
  const float* X    = z == 0 ? xq : z == 1 ? xk : xv;
  const h16*   Wz   = WT + (size_t)z * D_MODEL * D_MODEL;
  const float* bias = z == 0 ? bq : z == 1 ? bk : bv;
  const int m0 = blockIdx.x * 256 + wave * 64;
  const int n0 = blockIdx.y * 64;

  v8f acc[4][4];
#pragma unroll
  for (int i = 0; i < 4; ++i)
#pragma unroll
    for (int j = 0; j < 4; ++j) acc[i][j] = vzero();

  for (int k0 = 0; k0 < D_MODEL; k0 += 32) {
    v16h a[4], b[4];
#pragma unroll
    for (int i = 0; i < 4; ++i) a[i] = load_a32(X, D_MODEL, m0 + 16 * i, k0, lane);
#pragma unroll
    for (int j = 0; j < 4; ++j) b[j] = load_b16(Wz, D_MODEL, n0 + 16 * j, k0, lane);
#pragma unroll
    for (int i = 0; i < 4; ++i)
#pragma unroll
      for (int j = 0; j < 4; ++j) acc[i][j] = wmma_f16(a[i], b[j], acc[i][j]);
  }

  const int ln = lane & 15;
  const int rsel = (lane >> 4) * 8;   // lanes>=16 hold rows M=v+8
#pragma unroll
  for (int j = 0; j < 4; ++j) {
    const int c = n0 + 16 * j + ln;
    const int h = c >> 6, dk = c & 63;
    const float bval = bias[c];
#pragma unroll
    for (int i = 0; i < 4; ++i) {
#pragma unroll
      for (int v = 0; v < 8; ++v) {
        const int r = m0 + 16 * i + v + rsel;
        const int b_ = r >> 11, s = r & 2047;
        const size_t bh = (size_t)(b_ * N_HEAD + h);
        const h16 val = (h16)(acc[i][j][v] + bval);
        if (z == 0)      Qh[(bh * SEQ + s) * D_K + dk] = val;
        else if (z == 1) Kh[(bh * SEQ + s) * D_K + dk] = val;
        else             VhT[(bh * D_K + dk) * SEQ + s] = val;   // transposed
      }
    }
  }
}

// ---------------------------------------------------------------------------
// Kernel 2: streaming sigmoid attention. Wave owns 64 query rows x full d_k=64.
// ctx += sigmoid(Q K^T / 8) @ V, streamed over t in chunks of 32.
__global__ __launch_bounds__(64) void attn_kernel(const h16* Qh, const h16* Kh,
                                                  const h16* VhT, h16* ctx16) {
  __shared__ h16 smem[2][64 * LDSTR];
  const int lane = threadIdx.x & 31;
  const int wave = threadIdx.x >> 5;
  const int bh = blockIdx.y;
  const int s0 = (blockIdx.x * 2 + wave) * 64;
  const h16* Q  = Qh  + (size_t)bh * SEQ * D_K;
  const h16* K  = Kh  + (size_t)bh * SEQ * D_K;
  const h16* Vt = VhT + (size_t)bh * D_K * SEQ;
  h16* sm = smem[wave];
  const int ln = lane & 15;
  const int rsel = (lane >> 4) * 8;

  // Q fragments stay resident: 4 m-tiles x 2 k-steps over d_k=64
  v16h qf[4][2];
#pragma unroll
  for (int i = 0; i < 4; ++i)
#pragma unroll
    for (int kd = 0; kd < 2; ++kd)
      qf[i][kd] = load_a16(Q, D_K, s0 + 16 * i, 32 * kd, lane);

  v8f ctx[4][4];
#pragma unroll
  for (int i = 0; i < 4; ++i)
#pragma unroll
    for (int n = 0; n < 4; ++n) ctx[i][n] = vzero();

  for (int t0 = 0; t0 < SEQ; t0 += 32) {
    // ---- scores tile S[64 x 32] = Q K^T (K-dim = d_k = 64) ----
    v8f sc[4][2];
#pragma unroll
    for (int i = 0; i < 4; ++i)
#pragma unroll
      for (int j = 0; j < 2; ++j) sc[i][j] = vzero();
    v16h kf[2][2];
#pragma unroll
    for (int j = 0; j < 2; ++j)
#pragma unroll
      for (int kd = 0; kd < 2; ++kd)
        kf[j][kd] = load_b16(K, D_K, t0 + 16 * j, 32 * kd, lane);
#pragma unroll
    for (int kd = 0; kd < 2; ++kd)
#pragma unroll
      for (int i = 0; i < 4; ++i)
#pragma unroll
        for (int j = 0; j < 2; ++j) sc[i][j] = wmma_f16(qf[i][kd], kf[j][kd], sc[i][j]);

    // ---- sigmoid(x/8) and repack C-layout -> row-major [s][t] f16 in LDS ----
#pragma unroll
    for (int i = 0; i < 4; ++i)
#pragma unroll
      for (int j = 0; j < 2; ++j)
#pragma unroll
        for (int v = 0; v < 8; ++v) {
          const float x = sc[i][j][v] * 0.125f;           // 1/sqrt(d_k)
          const float a = 1.0f / (1.0f + __expf(-x));     // sigmoid, no row norm
          sm[(16 * i + v + rsel) * LDSTR + 16 * j + ln] = (h16)a;
        }
    asm volatile("s_wait_dscnt 0" ::: "memory");   // cross-lane LDS handoff

    // ---- ctx[64 x 64] += attn[64 x 32] @ V[32 x 64] ----
    v16h af[4], vf[4];
#pragma unroll
    for (int i = 0; i < 4; ++i) af[i] = load_a_lds(sm, 16 * i, lane);
#pragma unroll
    for (int n = 0; n < 4; ++n) vf[n] = load_b16(Vt, SEQ, 16 * n, t0, lane);
#pragma unroll
    for (int i = 0; i < 4; ++i)
#pragma unroll
      for (int n = 0; n < 4; ++n) ctx[i][n] = wmma_f16(af[i], vf[n], ctx[i][n]);
    asm volatile("s_wait_dscnt 0" ::: "memory");   // reads done before next overwrite
  }

  // ---- write ctx in merged-head layout [b*S + s][h*64 + d] as f16 ----
  const int b_ = bh >> 4, h = bh & 15;
#pragma unroll
  for (int i = 0; i < 4; ++i)
#pragma unroll
    for (int n = 0; n < 4; ++n)
#pragma unroll
      for (int v = 0; v < 8; ++v) {
        const int s = s0 + 16 * i + v + rsel;
        const int c = h * 64 + 16 * n + ln;
        ctx16[((size_t)(b_ * SEQ + s)) * D_MODEL + c] = (h16)ctx[i][n][v];
      }
}

// ---------------------------------------------------------------------------
// Kernel 3: out = ctx16 @ Wo + bo, f32 output.
__global__ __launch_bounds__(128) void out_proj_kernel(const h16* ctx16, const h16* WoT,
                                                       const float* bo, float* out) {
  const int lane = threadIdx.x & 31;
  const int wave = threadIdx.x >> 5;
  const int m0 = blockIdx.x * 256 + wave * 64;
  const int n0 = blockIdx.y * 64;

  v8f acc[4][4];
#pragma unroll
  for (int i = 0; i < 4; ++i)
#pragma unroll
    for (int j = 0; j < 4; ++j) acc[i][j] = vzero();

  for (int k0 = 0; k0 < D_MODEL; k0 += 32) {
    v16h a[4], b[4];
#pragma unroll
    for (int i = 0; i < 4; ++i) a[i] = load_a16(ctx16, D_MODEL, m0 + 16 * i, k0, lane);
#pragma unroll
    for (int j = 0; j < 4; ++j) b[j] = load_b16(WoT, D_MODEL, n0 + 16 * j, k0, lane);
#pragma unroll
    for (int i = 0; i < 4; ++i)
#pragma unroll
      for (int j = 0; j < 4; ++j) acc[i][j] = wmma_f16(a[i], b[j], acc[i][j]);
  }

  const int ln = lane & 15;
  const int rsel = (lane >> 4) * 8;
#pragma unroll
  for (int j = 0; j < 4; ++j) {
    const int c = n0 + 16 * j + ln;
    const float bval = bo[c];
#pragma unroll
    for (int i = 0; i < 4; ++i)
#pragma unroll
      for (int v = 0; v < 8; ++v) {
        const int r = m0 + 16 * i + v + rsel;
        out[(size_t)r * D_MODEL + c] = acc[i][j][v] + bval;
      }
  }
}

// ---------------------------------------------------------------------------
extern "C" void kernel_launch(void* const* d_in, const int* in_sizes, int n_in,
                              void* d_out, int out_size, void* d_ws, size_t ws_size,
                              hipStream_t stream) {
  const float* q  = (const float*)d_in[0];
  const float* k  = (const float*)d_in[1];
  const float* v  = (const float*)d_in[2];
  const float* Wq = (const float*)d_in[3];
  const float* bq = (const float*)d_in[4];
  const float* Wk = (const float*)d_in[5];
  const float* bk = (const float*)d_in[6];
  const float* Wv = (const float*)d_in[7];
  const float* bv = (const float*)d_in[8];
  const float* Wo = (const float*)d_in[9];
  const float* bo = (const float*)d_in[10];

  // Workspace: 4x W^T (8MB) + Qh/Kh/VhT (24MB) + ctx16 (8MB) = 40MB of f16.
  h16* WT  = (h16*)d_ws;
  h16* Qh  = WT  + (size_t)4 * D_MODEL * D_MODEL;
  h16* Kh  = Qh  + (size_t)BATCH * N_HEAD * SEQ * D_K;
  h16* VhT = Kh  + (size_t)BATCH * N_HEAD * SEQ * D_K;
  h16* ctx = VhT + (size_t)BATCH * N_HEAD * SEQ * D_K;

  wcvt_kernel<<<dim3(D_MODEL * D_MODEL / 256, 1, 4), 256, 0, stream>>>(Wq, Wk, Wv, Wo, WT);
  qkv_proj_kernel<<<dim3(ROWS / 256, D_MODEL / 64, 3), 128, 0, stream>>>(
      q, k, v, WT, bq, bk, bv, Qh, Kh, VhT);
  attn_kernel<<<dim3(SEQ / 128, BATCH * N_HEAD), 64, 0, stream>>>(Qh, Kh, VhT, ctx);
  out_proj_kernel<<<dim3(ROWS / 256, D_MODEL / 64), 128, 0, stream>>>(
      ctx, WT + (size_t)3 * D_MODEL * D_MODEL, bo, (float*)d_out);
}
